// ContextualLoss_38594576121983
// MI455X (gfx1250) — matrix operand order repaired and checked by reading, compiled, not run on hardware
//
#include <hip/hip_runtime.h>
#include <math.h>

// ---------------------------------------------------------------------------
// Types / helpers
// ---------------------------------------------------------------------------
typedef unsigned short u16;
typedef __attribute__((ext_vector_type(16))) __bf16 v16bf;
typedef __attribute__((ext_vector_type(8)))  __bf16 v8bf;
typedef __attribute__((ext_vector_type(8)))  float  v8f;

union Frag { v16bf v; v8bf h[2]; };
union BPack { u16 s[8]; v8bf v; };

__device__ __forceinline__ u16 f2bf(float f) {
  union { float f; unsigned u; } a; a.f = f;
  unsigned r = a.u + 0x7FFFu + ((a.u >> 16) & 1u);   // round-to-nearest-even
  return (u16)(r >> 16);
}
__device__ __forceinline__ float bf2f(u16 h) {
  union { unsigned u; float f; } a; a.u = ((unsigned)h) << 16;
  return a.f;
}
__device__ __forceinline__ void atomicMaxF(float* addr, float val) {
  unsigned* p = (unsigned*)addr;
  unsigned old = *p, assumed;
  do {
    assumed = old;
    if (__uint_as_float(assumed) >= val) break;
    old = atomicCAS(p, assumed, __float_as_uint(val));
  } while (old != assumed);
}

// --- CDNA5 async global->LDS copy (ASYNCcnt path, ISA 08_async_tensor §4) ---
__device__ __forceinline__ unsigned lds_off(void* p) {
  return (unsigned)(unsigned long long)(__attribute__((address_space(3))) void*)p;
}
__device__ __forceinline__ void async_ld_b128(unsigned ldsOff, const void* g) {
  asm volatile("global_load_async_to_lds_b128 %0, %1, off"
               :: "v"(ldsOff), "v"((unsigned long long)(size_t)g)
               : "memory");
}
__device__ __forceinline__ void wait_async0() {
  asm volatile("s_wait_asynccnt 0x0" ::: "memory");
}

// ---------------------------------------------------------------------------
// 1. Preprocess: (x - mean)/std, f32 NCHW -> bf16 NCHW   [2][3][256][256]
// ---------------------------------------------------------------------------
__global__ void k_preprocess(const float* __restrict__ in, u16* __restrict__ out, int total) {
  int i = blockIdx.x * 256 + threadIdx.x;
  if (i >= total) return;
  int c = (i >> 16) % 3;  // 256*256 = 65536 per plane
  float mean = (c == 0) ? 0.485f : (c == 1) ? 0.456f : 0.406f;
  float stdv = (c == 0) ? 0.229f : (c == 1) ? 0.224f : 0.225f;
  out[i] = f2bf((in[i] - mean) / stdv);
}

// ---------------------------------------------------------------------------
// 2. Weight prep: f32 OIHW -> bf16 [Cout][Kpad], K = cin*9 + kh*3 + kw
// ---------------------------------------------------------------------------
__global__ void k_wprep(const float* __restrict__ w, u16* __restrict__ wb,
                        int CinK, int Kpad, int total) {
  int i = blockIdx.x * 256 + threadIdx.x;
  if (i >= total) return;
  int k = i % Kpad;
  int co = i / Kpad;
  wb[i] = (k < CinK) ? f2bf(w[co * CinK + k]) : (u16)0;
}

// ---------------------------------------------------------------------------
// 3. Conv3x3 + bias + ReLU, implicit GEMM via v_wmma_f32_16x16x32_bf16.
//    Block tile 64(cout) x 64(pixel); 8 waves as 4x2, 2 accumulators/wave.
//    Weights staged via async global->LDS; patch gather uses a per-block
//    precomputed (cin,kh,kw) LDS table. grid: (HW/64, Cout/64, N=2).
// ---------------------------------------------------------------------------
__global__ void __launch_bounds__(256) k_conv_wmma(
    const u16* __restrict__ in, const u16* __restrict__ wgt,
    const float* __restrict__ bias, u16* __restrict__ out,
    int Cin, int Cout, int H, int W, int Kpad) {
  __shared__ __align__(16) u16 As[64 * 32];   // weights tile  [m][k]      4 KB
  __shared__ __align__(16) u16 Bs[64 * 32];   // patch tile    [pixel][k]  4 KB
  __shared__ int ktab[2304];                  // packed (cin,kh,kw) per k  9 KB

  const int tid  = threadIdx.x;
  const int lane = tid & 31;
  const int wv   = tid >> 5;
  const int mi   = wv & 3;          // cout subtile 0..3
  const int ni   = wv >> 2;         // pixel group 0..1 (each covers 2 subtiles)
  const int lr   = lane & 15;
  const int half = lane >> 4;
  const int kbA  = half * 16;       // A fragment K run base   (contiguous k = 16*half + s)
  const int kbB  = half * 16;       // B fragment K run base   (contiguous k = 16*half + s)
  const int HW   = H * W;
  const int pixBase  = blockIdx.x * 64;
  const int coutBase = blockIdx.y * 64;
  const int nImg     = blockIdx.z;

  // Build k -> (cin, kh, kw) table once per block.
  for (int k = tid; k < Kpad; k += 256) {
    int cin = k / 9, t9 = k - cin * 9;
    ktab[k] = (cin < Cin) ? (cin | ((t9 / 3) << 8) | ((t9 % 3) << 10)) : -1;
  }

  v8f acc0, acc1;
#pragma unroll
  for (int r = 0; r < 8; ++r) { acc0[r] = 0.f; acc1[r] = 0.f; }

  // A stage assignment: 64x32 halfs, one async b128 per thread
  const int arow = tid >> 2;         // 0..63
  const int acol = (tid & 3) * 8;    // 0,8,16,24
  const unsigned aLds = lds_off(&As[arow * 32 + acol]);
  // B gather assignment: 64 pixels x 32 k, 8 halfs (one b128 LDS store)/thread
  const int brow  = tid >> 2;        // 0..63 (pixel in tile)
  const int bcol8 = (tid & 3) * 8;   // 0,8,16,24
  const int pix = pixBase + brow;
  const int ph  = pix / W;
  const int pw  = pix - ph * W;

  __syncthreads();  // ktab visible

  for (int k0 = 0; k0 < Kpad; k0 += 32) {
    // Stage A (weights) via CDNA5 async global->LDS
    async_ld_b128(aLds, &wgt[(size_t)(coutBase + arow) * Kpad + k0 + acol]);
    // Stage B (patches): table-driven gather, one b128 LDS store
    BPack bt;
#pragma unroll
    for (int j = 0; j < 8; ++j) {
      int pk = ktab[k0 + bcol8 + j];
      u16 v = 0;
      if (pk >= 0) {
        int cin = pk & 255;
        int h = ph + (((pk >> 8) & 3) - 1);
        int w = pw + (((pk >> 10) & 3) - 1);
        if ((unsigned)h < (unsigned)H && (unsigned)w < (unsigned)W)
          v = in[(size_t)(nImg * Cin + cin) * HW + h * W + w];
      }
      bt.s[j] = v;
    }
    *(v8bf*)&Bs[brow * 32 + bcol8] = bt.v;
    wait_async0();
    __syncthreads();

    Frag a, b0, b1;
    a.h[0]  = *(const v8bf*)&As[(mi * 16 + lr) * 32 + kbA];
    a.h[1]  = *(const v8bf*)&As[(mi * 16 + lr) * 32 + kbA + 8];
    b0.h[0] = *(const v8bf*)&Bs[(ni * 32 + lr) * 32 + kbB];
    b0.h[1] = *(const v8bf*)&Bs[(ni * 32 + lr) * 32 + kbB + 8];
    b1.h[0] = *(const v8bf*)&Bs[(ni * 32 + 16 + lr) * 32 + kbB];
    b1.h[1] = *(const v8bf*)&Bs[(ni * 32 + 16 + lr) * 32 + kbB + 8];
    acc0 = __builtin_amdgcn_wmma_f32_16x16x32_bf16(false, a.v, false, b0.v,
                                                   (short)0, acc0, false, false);
    acc1 = __builtin_amdgcn_wmma_f32_16x16x32_bf16(false, a.v, false, b1.v,
                                                   (short)0, acc1, false, false);
    __syncthreads();
  }

  // Epilogue: bias + relu. C layout: VGPR r -> M = r + 8*half, N = lr
  const int p0 = pixBase + ni * 32 + lr;
#pragma unroll
  for (int r = 0; r < 8; ++r) {
    int m = coutBase + mi * 16 + half * 8 + r;
    float b = bias[m];
    float v0 = acc0[r] + b; v0 = v0 > 0.f ? v0 : 0.f;
    float v1 = acc1[r] + b; v1 = v1 > 0.f ? v1 : 0.f;
    size_t base = (size_t)(nImg * Cout + m) * HW;
    out[base + p0]      = f2bf(v0);
    out[base + p0 + 16] = f2bf(v1);
  }
}

// ---------------------------------------------------------------------------
// 4. 2x2 max pool (bf16)
// ---------------------------------------------------------------------------
__global__ void k_pool(const u16* __restrict__ in, u16* __restrict__ out,
                       int C, int H, int W, int total) {
  int i = blockIdx.x * 256 + threadIdx.x;
  if (i >= total) return;
  int Wo = W >> 1, Ho = H >> 1;
  int wo = i % Wo; int t = i / Wo;
  int ho = t % Ho; t /= Ho;
  int c = t % C;   int n = t / C;
  const u16* p = &in[(size_t)(n * C + c) * H * W + (ho * 2) * W + wo * 2];
  float a = bf2f(p[0]), b = bf2f(p[1]), cc = bf2f(p[W]), d = bf2f(p[W + 1]);
  out[i] = f2bf(fmaxf(fmaxf(a, b), fmaxf(cc, d)));
}

// ---------------------------------------------------------------------------
// 5. Per-channel mean of fy over (n,h,w).  fy: [2][256][4096] bf16 -> mu[256]
// ---------------------------------------------------------------------------
__global__ void k_chanmean(const u16* __restrict__ fy, float* __restrict__ mu) {
  __shared__ float red[256];
  int c = blockIdx.x;
  float s = 0.f;
  for (int idx = threadIdx.x; idx < 2 * 4096; idx += 256) {
    int n = idx >> 12, p = idx & 4095;
    s += bf2f(fy[(size_t)(n * 256 + c) * 4096 + p]);
  }
  red[threadIdx.x] = s; __syncthreads();
  for (int st = 128; st > 0; st >>= 1) {
    if (threadIdx.x < st) red[threadIdx.x] += red[threadIdx.x + st];
    __syncthreads();
  }
  if (threadIdx.x == 0) mu[c] = red[0] * (1.f / 8192.f);
}

// ---------------------------------------------------------------------------
// 6. Center + channel-L2-normalize, write transposed [n*4096 + p][256] bf16
// ---------------------------------------------------------------------------
__global__ void __launch_bounds__(256) k_featnorm(
    const u16* __restrict__ feat, const float* __restrict__ mu,
    u16* __restrict__ outT) {
  int wv = threadIdx.x >> 5, lane = threadIdx.x & 31;
  int pos = blockIdx.x * 8 + wv;     // 0..8191
  int n = pos >> 12, p = pos & 4095;
  float v[8], ss = 0.f;
#pragma unroll
  for (int j = 0; j < 8; ++j) {
    int c = lane + j * 32;
    float x = bf2f(feat[(size_t)(n * 256 + c) * 4096 + p]) - mu[c];
    v[j] = x; ss += x * x;
  }
#pragma unroll
  for (int m = 16; m >= 1; m >>= 1) ss += __shfl_xor(ss, m, 32);
  float inv = 1.f / fmaxf(sqrtf(ss), 1e-12f);
#pragma unroll
  for (int j = 0; j < 8; ++j) {
    int c = lane + j * 32;
    outT[(size_t)pos * 256 + c] = f2bf(v[j] * inv);
  }
}

// ---------------------------------------------------------------------------
// 7. Init reduction buffers
// ---------------------------------------------------------------------------
__global__ void k_redinit(float* rowmax, float* rowsum, float* colmax) {
  int i = blockIdx.x * 256 + threadIdx.x;
  if (i < 2 * 4096) { rowmax[i] = -1e30f; rowsum[i] = 0.f; colmax[i] = 0.f; }
}

// ---------------------------------------------------------------------------
// 8. Contextual-loss GEMM passes over S = A x B^T (K=256), M-tile 64/block,
//    looping all N; panels staged with async global->LDS.
//    PASS 0: rowmax(S); PASS 1: rowsum of w; PASS 2 (A=yn,B=xn): colmax w/rowsum.
//    grid (4096/64, 2); block 256 = 8 waves (4x2).
// ---------------------------------------------------------------------------
template <int PASS>
__global__ void __launch_bounds__(256) k_cxpass(
    const u16* __restrict__ Abuf, const u16* __restrict__ Bbuf,
    const float* __restrict__ rowmax, const float* __restrict__ rowsum,
    float* __restrict__ outred) {
  __shared__ __align__(16) u16 Al[64 * 256];  // 32 KB, whole K panel of A
  __shared__ __align__(16) u16 Bl[32 * 256];  // 16 KB, per-chunk B panel
  const int tid  = threadIdx.x;
  const int lane = tid & 31;
  const int wv   = tid >> 5;
  const int mi   = wv & 3, ni = wv >> 2;
  const int lr   = lane & 15, half = lane >> 4;
  const int kbA  = half * 16, kbB = half * 16;
  const int mBase = blockIdx.x * 64;
  const int n4096 = blockIdx.y * 4096;
  const u16* A = Abuf + (size_t)blockIdx.y * 4096 * 256;
  const u16* B = Bbuf + (size_t)blockIdx.y * 4096 * 256;

  {  // stage A panel 64x256 via async copies (8 x b128 per thread)
    int row = tid >> 2, colb = (tid & 3) * 64;
#pragma unroll
    for (int j = 0; j < 8; ++j)
      async_ld_b128(lds_off(&Al[row * 256 + colb + j * 8]),
                    &A[(size_t)(mBase + row) * 256 + colb + j * 8]);
  }

  float racc[8];
#pragma unroll
  for (int r = 0; r < 8; ++r) racc[r] = (PASS == 1) ? 0.f : -1e30f;

  float rminv[8];
  if (PASS == 1) {
#pragma unroll
    for (int r = 0; r < 8; ++r) {
      int m = mBase + mi * 16 + half * 8 + r;
      rminv[r] = (1.f - rowmax[n4096 + m]) + 1e-5f;  // dist.min + 1e-5
    }
  }

  const int brow = tid >> 3, bcolb = (tid & 7) * 32;

  for (int jc = 0; jc < 4096; jc += 32) {
    __syncthreads();  // all reads of previous Bl done before overwrite
#pragma unroll
    for (int j = 0; j < 4; ++j)
      async_ld_b128(lds_off(&Bl[brow * 256 + bcolb + j * 8]),
                    &B[(size_t)(jc + brow) * 256 + bcolb + j * 8]);
    wait_async0();
    __syncthreads();

    v8f c;
#pragma unroll
    for (int r = 0; r < 8; ++r) c[r] = 0.f;
#pragma unroll
    for (int ks = 0; ks < 8; ++ks) {
      int k0 = ks * 32;
      Frag a, b;
      a.h[0] = *(const v8bf*)&Al[(mi * 16 + lr) * 256 + k0 + kbA];
      a.h[1] = *(const v8bf*)&Al[(mi * 16 + lr) * 256 + k0 + kbA + 8];
      b.h[0] = *(const v8bf*)&Bl[(ni * 16 + lr) * 256 + k0 + kbB];
      b.h[1] = *(const v8bf*)&Bl[(ni * 16 + lr) * 256 + k0 + kbB + 8];
      c = __builtin_amdgcn_wmma_f32_16x16x32_bf16(false, a.v, false, b.v,
                                                  (short)0, c, false, false);
    }

    if (PASS == 0) {
#pragma unroll
      for (int r = 0; r < 8; ++r) racc[r] = fmaxf(racc[r], c[r]);
    } else if (PASS == 1) {
#pragma unroll
      for (int r = 0; r < 8; ++r) {
        float dist = 1.f - c[r];
        racc[r] += __expf((1.f - dist / rminv[r]) * 2.f);  // 1/band_width = 2
      }
    } else {
      int icol = jc + ni * 16 + lr;                 // original i index
      float rmin = (1.f - rowmax[n4096 + icol]) + 1e-5f;
      float inv  = 1.f / rowsum[n4096 + icol];
#pragma unroll
      for (int r = 0; r < 8; ++r) {
        float dist = 1.f - c[r];
        float w = __expf((1.f - dist / rmin) * 2.f);
        racc[r] = fmaxf(racc[r], w * inv);
      }
    }
  }

  // reduce across the 16-lane group holding the same C-tile rows
#pragma unroll
  for (int r = 0; r < 8; ++r) {
    float v = racc[r];
#pragma unroll
    for (int m = 1; m < 16; m <<= 1) {
      float o = __shfl_xor(v, m, 32);
      v = (PASS == 1) ? (v + o) : fmaxf(v, o);
    }
    racc[r] = v;
  }
  if (lr == 0) {
#pragma unroll
    for (int r = 0; r < 8; ++r) {
      int m = mBase + mi * 16 + half * 8 + r;
      if (PASS == 1) atomicAdd(&outred[n4096 + m], racc[r]);
      else           atomicMaxF(&outred[n4096 + m], racc[r]);
    }
  }
}

// ---------------------------------------------------------------------------
// 9. Final: loss = mean_n( -log( mean_j colmax[n][j] + 1e-5 ) )
// ---------------------------------------------------------------------------
__global__ void k_final(const float* __restrict__ colmax, float* __restrict__ out) {
  __shared__ float red[256];
  float loss = 0.f;
  for (int n = 0; n < 2; ++n) {
    float s = 0.f;
    for (int j = threadIdx.x; j < 4096; j += 256) s += colmax[n * 4096 + j];
    red[threadIdx.x] = s; __syncthreads();
    for (int st = 128; st > 0; st >>= 1) {
      if (threadIdx.x < st) red[threadIdx.x] += red[threadIdx.x + st];
      __syncthreads();
    }
    if (threadIdx.x == 0) loss += -logf(red[0] * (1.f / 4096.f) + 1e-5f);
    __syncthreads();
  }
  if (threadIdx.x == 0) out[0] = loss * 0.5f;
}

// ---------------------------------------------------------------------------
// Host orchestration
// ---------------------------------------------------------------------------
extern "C" void kernel_launch(void* const* d_in, const int* in_sizes, int n_in,
                              void* d_out, int out_size, void* d_ws, size_t ws_size,
                              hipStream_t stream) {
  (void)in_sizes; (void)n_in; (void)out_size; (void)ws_size;
  const float* x = (const float*)d_in[0];
  const float* y = (const float*)d_in[1];
  const float* Wt[8]; const float* Bi[8];
  for (int i = 0; i < 8; ++i) { Wt[i] = (const float*)d_in[2 + i]; Bi[i] = (const float*)d_in[10 + i]; }

  char* ws = (char*)d_ws;
  u16*   actA = (u16*)(ws + 0);                       // 16 MiB
  u16*   actB = (u16*)(ws + (16u << 20));             // 16 MiB
  u16*   wbuf = (u16*)(ws + (32u << 20));             // <= 1.2 MiB used
  u16*   fxb  = (u16*)(ws + (34u << 20));             // 4 MiB
  u16*   fyb  = (u16*)(ws + (38u << 20));             // 4 MiB
  u16*   xnT  = (u16*)(ws + (42u << 20));             // 4 MiB
  u16*   ynT  = (u16*)(ws + (46u << 20));             // 4 MiB
  float* mu   = (float*)(ws + (50u << 20));
  float* rmax = (float*)(ws + (50u << 20) + 4096);
  float* rsum = (float*)(ws + (50u << 20) + 4096 + 32768);
  float* cmax = (float*)(ws + (50u << 20) + 4096 + 65536);

  const int CIN[8]  = {3, 64, 64, 128, 128, 256, 256, 256};
  const int COUT[8] = {64, 64, 128, 128, 256, 256, 256, 256};

  for (int img = 0; img < 2; ++img) {
    const float* src = (img == 0) ? x : y;
    u16* featOut = (img == 0) ? fxb : fyb;
    k_preprocess<<<(393216 + 255) / 256, 256, 0, stream>>>(src, actA, 393216);
    u16 *cur = actA, *nxt = actB;
    int H = 256, Wd = 256;
    for (int L = 0; L < 8; ++L) {
      int Cin = CIN[L], Cout = COUT[L];
      int CinK = Cin * 9, Kpad = ((CinK + 31) / 32) * 32;
      int wtot = Cout * Kpad;
      k_wprep<<<(wtot + 255) / 256, 256, 0, stream>>>(Wt[L], wbuf, CinK, Kpad, wtot);
      u16* outp = (L == 7) ? featOut : nxt;
      dim3 g(H * Wd / 64, Cout / 64, 2);
      k_conv_wmma<<<g, 256, 0, stream>>>(cur, wbuf, Bi[L], outp, Cin, Cout, H, Wd, Kpad);
      if (L == 1 || L == 3) {
        int tot = 2 * Cout * (H / 2) * (Wd / 2);
        k_pool<<<(tot + 255) / 256, 256, 0, stream>>>(nxt, cur, Cout, H, Wd, tot);
        H >>= 1; Wd >>= 1;          // pooled result in cur -> next layer input
      } else {
        u16* t = cur; cur = nxt; nxt = t;
      }
    }
  }

  k_chanmean<<<256, 256, 0, stream>>>(fyb, mu);
  k_featnorm<<<1024, 256, 0, stream>>>(fxb, mu, xnT);
  k_featnorm<<<1024, 256, 0, stream>>>(fyb, mu, ynT);
  k_redinit<<<32, 256, 0, stream>>>(rmax, rsum, cmax);
  dim3 gg(64, 2);
  k_cxpass<0><<<gg, 256, 0, stream>>>(xnT, ynT, rmax, rsum, rmax);
  k_cxpass<1><<<gg, 256, 0, stream>>>(xnT, ynT, rmax, rsum, rsum);
  k_cxpass<2><<<gg, 256, 0, stream>>>(ynT, xnT, rmax, rsum, cmax);
  k_final<<<1, 256, 0, stream>>>(cmax, (float*)d_out);
}
